// SpMiddleAlinPreFramesNoDownsampleXYMultiStep_21139829031373
// MI455X (gfx1250) — matrix operand order repaired and checked by reading, not compiled, not run on hardware
//
#include <hip/hip_runtime.h>
#include <hip/hip_bf16.h>

typedef __attribute__((ext_vector_type(16))) _Float16 v16h;
typedef __attribute__((ext_vector_type(8)))  float    v8f;

#define BN_EPS 1e-3f

// ---------------------------------------------------------------------------
// Pack f32 weights [K, Ci, Co] into CDNA5 WMMA B-fragments (f16).
// Fragment (k, cit, cot): 32 lanes x 16 halfs; element e of lane L is
//   w[k][cit*32 + e + (e>=8 ? 8 : 0) + (L>=16 ? 8 : 0)][cot*16 + (L&15)]   (0 if ci >= Ci)
// Fragment order: ((k*ciT + cit)*coT + cot), 1 KB each, contiguous.
// ---------------------------------------------------------------------------
__global__ void pack_w_kernel(const float* __restrict__ w, _Float16* __restrict__ p,
                              int K, int Ci, int Co) {
    int coT = Co / 16;
    int ciT = (Ci + 31) / 32;
    int bid = blockIdx.x;
    int cot = bid % coT;
    int tmp = bid / coT;
    int cit = tmp % ciT;
    int k   = tmp / ciT;
    if (k >= K) return;
    int lane = threadIdx.x;           // 0..31
    int n    = lane & 15;             // output-channel within tile
    int kofs = (lane < 16) ? 0 : 8;   // K offset for upper half-wave
    _Float16* dst = p + (((size_t)bid) * 32 + lane) * 16;
#pragma unroll
    for (int e = 0; e < 16; ++e) {
        int ci = cit * 32 + kofs + e + (e >= 8 ? 8 : 0);
        float v = (ci < Ci) ? w[((size_t)k * Ci + ci) * Co + cot * 16 + n] : 0.0f;
        dst[e] = (_Float16)v;
    }
}

// ---------------------------------------------------------------------------
// Gather-GEMM + BN + ReLU (+ optional broadcast temporal add) via WMMA,
// specialized on (CI, CO, KT). One wave computes a 16(M) x CO output stripe:
// the gathered A fragment is reused across all CO/16 B tiles (4 independent
// WMMA chains for CO=64). 4 waves per block = 4 consecutive M tiles.
//   out[m, co] = relu( bn( sum_k sum_ci f[idx[m,k], ci] * w[k,ci,co] + a[co]*s ) )
// idx entry < 0 (or m >= M) contributes zeros. bnp: [g, b, mean, var] x CO.
// ---------------------------------------------------------------------------
template <int CI, int CO, int KT>
__global__ void gconv_wmma_kernel(const float* __restrict__ fin,
                                  const int*   __restrict__ idx,
                                  const _Float16* __restrict__ wpack,
                                  const float* __restrict__ bnp,
                                  const float* __restrict__ aadd, float ascale,
                                  float* __restrict__ fout, int M) {
    constexpr int ciT = (CI + 31) / 32;
    constexpr int coT = CO / 16;

    const int wave = threadIdx.x >> 5;               // 0..3
    const int lane = threadIdx.x & 31;               // 0..31
    const int mt   = blockIdx.x * 4 + wave;          // M tile
    if (mt * 16 >= M) return;                        // wave-uniform: EXEC stays all-1s
    const int row  = mt * 16 + (lane & 15);
    const bool hi  = (lane >= 16);
    const int aofs = hi ? 8 : 0;                     // A-fragment K offset per half-wave

    v8f acc[coT];
#pragma unroll
    for (int c = 0; c < coT; ++c) acc[c] = (v8f){};

#pragma unroll 3
    for (int k = 0; k < KT; ++k) {
        int g = (row < M) ? idx[(size_t)row * KT + k] : -1;
        const float* src = (g >= 0) ? (fin + (size_t)g * CI) : nullptr;
#pragma unroll
        for (int cit = 0; cit < ciT; ++cit) {
            const int cibase = cit * 32;
            v16h A;
            if (src) {
                // low 8 K-values: ci = cibase + aofs + [0..7]  (always in range)
                const float4* p0 = (const float4*)(src + cibase + aofs);
                float4 x0 = p0[0], x1 = p0[1];
                A[0] = (_Float16)x0.x; A[1] = (_Float16)x0.y;
                A[2] = (_Float16)x0.z; A[3] = (_Float16)x0.w;
                A[4] = (_Float16)x1.x; A[5] = (_Float16)x1.y;
                A[6] = (_Float16)x1.z; A[7] = (_Float16)x1.w;
                if constexpr (CI >= 32) {
                    // high 8 K-values: ci = cibase + 16 + aofs + [0..7]
                    const float4* p1 = (const float4*)(src + cibase + 16 + aofs);
                    float4 y0 = p1[0], y1 = p1[1];
                    A[8]  = (_Float16)y0.x; A[9]  = (_Float16)y0.y;
                    A[10] = (_Float16)y0.z; A[11] = (_Float16)y0.w;
                    A[12] = (_Float16)y1.x; A[13] = (_Float16)y1.y;
                    A[14] = (_Float16)y1.z; A[15] = (_Float16)y1.w;
                } else {
                    // CI == 16: upper half of the K=32 chunk is zero padding
#pragma unroll
                    for (int e = 8; e < 16; ++e) A[e] = (_Float16)0.0f;
                }
            } else {
#pragma unroll
                for (int e = 0; e < 16; ++e) A[e] = (_Float16)0.0f;
            }
            const _Float16* bbase =
                wpack + (((size_t)k * ciT + cit) * coT * 32 + lane) * 16;
            __builtin_prefetch(bbase + (size_t)ciT * coT * 32 * 16, 0, 1);  // next tap
#pragma unroll
            for (int c = 0; c < coT; ++c) {
                v16h B = *(const v16h*)(bbase + (size_t)c * 32 * 16);
                acc[c] = __builtin_amdgcn_wmma_f32_16x16x32_f16(
                    false, A, false, B, (short)0, acc[c], false, false);
            }
        }
    }

    // Epilogue: temporal add + BN + ReLU. C/D layout: VGPR r -> row r (+8 hi half).
#pragma unroll
    for (int c = 0; c < coT; ++c) {
        const int col = c * 16 + (lane & 15);
        float g_ = bnp[col], b_ = bnp[CO + col];
        float m_ = bnp[2 * CO + col], v_ = bnp[3 * CO + col];
        float sc = g_ * rsqrtf(v_ + BN_EPS);
        float av = aadd ? aadd[col] * ascale : 0.0f;
#pragma unroll
        for (int r = 0; r < 8; ++r) {
            int orow = mt * 16 + r + (hi ? 8 : 0);
            if (orow < M) {
                float val = acc[c][r] + av;
                val = (val - m_) * sc + b_;
                fout[(size_t)orow * CO + col] = fmaxf(val, 0.0f);
            }
        }
    }
}

// ---------------------------------------------------------------------------
// accum[c] += sum_rows( relu(bn_ad(row @ w1)) @ w2 )[c]    (64 channels)
// Caller scales by 1/(10*M) when consuming. accum must be pre-zeroed.
// ---------------------------------------------------------------------------
__global__ void adapt_mean_kernel(const float* __restrict__ s, int M,
                                  const float* __restrict__ w1,
                                  const float* __restrict__ w2,
                                  const float* __restrict__ bnp,
                                  float* __restrict__ accum) {
    __shared__ float row[64];
    __shared__ float h[64];
    int c = threadIdx.x;  // 0..63
    float g_ = bnp[c], b_ = bnp[64 + c], m_ = bnp[128 + c], v_ = bnp[192 + c];
    float sc = g_ * rsqrtf(v_ + BN_EPS);
    float sum = 0.0f;
    for (int r = blockIdx.x; r < M; r += gridDim.x) {
        row[c] = s[(size_t)r * 64 + c];
        __syncthreads();
        float acc = 0.0f;
#pragma unroll 8
        for (int i = 0; i < 64; ++i) acc += row[i] * w1[i * 64 + c];
        h[c] = fmaxf((acc - m_) * sc + b_, 0.0f);
        __syncthreads();
        float y = 0.0f;
#pragma unroll 8
        for (int i = 0; i < 64; ++i) y += h[i] * w2[i * 64 + c];
        sum += y;
        __syncthreads();
    }
    atomicAdd(&accum[c], sum);
}

__global__ void zero_f32_kernel(float* __restrict__ p, size_t n) {
    size_t i = (size_t)blockIdx.x * blockDim.x + threadIdx.x;
    size_t st = (size_t)gridDim.x * blockDim.x;
    for (; i < n; i += st) p[i] = 0.0f;
}

// out[t, c*4+z, y, x] = x11[m, c]  with coords3[m] = (z, y, x); grid (4,320,320)
__global__ void scatter_out_kernel(const float* __restrict__ x,
                                   const int* __restrict__ coords,
                                   float* __restrict__ out, int M, int t) {
    int m = blockIdx.x;
    int c = threadIdx.x;  // 0..63
    if (m >= M) return;
    int z = coords[m * 3 + 0], y = coords[m * 3 + 1], xx = coords[m * 3 + 2];
    size_t o = ((((size_t)t * 256) + (size_t)c * 4 + z) * 320 + y) * 320 + xx;
    out[o] = x[(size_t)m * 64 + c];
}

// ---------------------------------------------------------------------------
extern "C" void kernel_launch(void* const* d_in, const int* in_sizes, int n_in,
                              void* d_out, int out_size, void* d_ws, size_t ws_size,
                              hipStream_t stream) {
    (void)n_in; (void)ws_size;
    const float* vf       = (const float*)d_in[0];
    const float* w_subm1  = (const float*)d_in[1];
    const float* w_subm2  = (const float*)d_in[2];
    const float* w_spcv1  = (const float*)d_in[3];
    const float* w_subm3  = (const float*)d_in[4];
    const float* w_subm4  = (const float*)d_in[5];
    const float* w_subm5  = (const float*)d_in[6];
    const float* w_spcv2  = (const float*)d_in[7];
    const float* w_subm6  = (const float*)d_in[8];
    const float* w_subm7  = (const float*)d_in[9];
    const float* w_subm8  = (const float*)d_in[10];
    const float* w_spcv3  = (const float*)d_in[11];
    const float* w_ad1    = (const float*)d_in[12];
    const float* w_ad2    = (const float*)d_in[13];
    const float* bn32     = (const float*)d_in[14];  // [2,4,32]
    const float* bn64     = (const float*)d_in[15];  // [9,4,64]
    const float* bn_ad    = (const float*)d_in[16];  // [4,64]
    const int* nbr0    = (const int*)d_in[17];
    const int* idx1    = (const int*)d_in[18];
    const int* nbr1    = (const int*)d_in[19];
    const int* idx2    = (const int*)d_in[20];
    const int* nbr2    = (const int*)d_in[21];
    const int* idx3    = (const int*)d_in[22];
    const int* coords3 = (const int*)d_in[23];

    const int M0 = in_sizes[17] / 27;
    const int M1 = in_sizes[18] / 27;
    const int M2 = in_sizes[20] / 27;
    const int M3 = in_sizes[22] / 3;
    const int T  = in_sizes[0] / (M0 * 16);

    // ---- workspace bump allocator ----
    size_t off = 0;
    auto alloc = [&](size_t bytes) -> char* {
        char* p = (char*)d_ws + off;
        off += (bytes + 255) & ~(size_t)255;
        return p;
    };
    auto fragCount = [](int K, int Ci, int Co) {
        return (size_t)K * ((Ci + 31) / 32) * (Co / 16);
    };
    auto allocPack = [&](int K, int Ci, int Co) {
        return (_Float16*)alloc(fragCount(K, Ci, Co) * 32 * 16 * sizeof(_Float16));
    };

    _Float16* p_subm1 = allocPack(27, 16, 32);
    _Float16* p_subm2 = allocPack(27, 32, 32);
    _Float16* p_spcv1 = allocPack(27, 32, 64);
    _Float16* p_subm3 = allocPack(27, 64, 64);
    _Float16* p_subm4 = allocPack(27, 64, 64);
    _Float16* p_subm5 = allocPack(27, 64, 64);
    _Float16* p_spcv2 = allocPack(27, 64, 64);
    _Float16* p_subm6 = allocPack(27, 64, 64);
    _Float16* p_subm7 = allocPack(27, 64, 64);
    _Float16* p_subm8 = allocPack(27, 64, 64);
    _Float16* p_spcv3 = allocPack(3, 64, 64);

    float* xA  = (float*)alloc((size_t)M0 * 32 * 4);
    float* xB  = (float*)alloc((size_t)M0 * 32 * 4);
    float* x3  = (float*)alloc((size_t)M1 * 64 * 4);
    float* yA  = (float*)alloc((size_t)M1 * 64 * 4);
    float* yB  = (float*)alloc((size_t)M1 * 64 * 4);
    float* x7  = (float*)alloc((size_t)M2 * 64 * 4);
    float* zA  = (float*)alloc((size_t)M2 * 64 * 4);
    float* zB  = (float*)alloc((size_t)M2 * 64 * 4);
    float* x11 = (float*)alloc((size_t)M3 * 64 * 4);
    float* a0  = (float*)alloc(64 * 4);
    float* a1  = (float*)alloc(64 * 4);
    float* a2  = (float*)alloc(64 * 4);

    // ---- pack weights to f16 B-fragments ----
    auto pack = [&](const float* w, _Float16* pw, int K, int Ci, int Co) {
        unsigned nfrag = (unsigned)fragCount(K, Ci, Co);
        pack_w_kernel<<<dim3(nfrag), dim3(32), 0, stream>>>(w, pw, K, Ci, Co);
    };
    pack(w_subm1, p_subm1, 27, 16, 32);
    pack(w_subm2, p_subm2, 27, 32, 32);
    pack(w_spcv1, p_spcv1, 27, 32, 64);
    pack(w_subm3, p_subm3, 27, 64, 64);
    pack(w_subm4, p_subm4, 27, 64, 64);
    pack(w_subm5, p_subm5, 27, 64, 64);
    pack(w_spcv2, p_spcv2, 27, 64, 64);
    pack(w_subm6, p_subm6, 27, 64, 64);
    pack(w_subm7, p_subm7, 27, 64, 64);
    pack(w_subm8, p_subm8, 27, 64, 64);
    pack(w_spcv3, p_spcv3, 3, 64, 64);

    // ---- zero dense output (only scattered coords get written) ----
    zero_f32_kernel<<<dim3(2048), dim3(256), 0, stream>>>((float*)d_out, (size_t)out_size);

    auto blocksFor = [](int M) { return dim3((unsigned)((M + 63) / 64)); };
    const dim3 blk(128);  // 4 waves, 4 M tiles

    const float s0scale = 1.0f / (10.0f * (float)M1);
    const float s1scale = 1.0f / (10.0f * (float)M2);
    const float s2scale = 1.0f / (10.0f * (float)M3);

    for (int t = 0; t < T; ++t) {
        const float* ft = vf + (size_t)t * M0 * 16;
        const float* A0 = (t > 0) ? a0 : nullptr;
        const float* A1 = (t > 0) ? a1 : nullptr;
        const float* A2 = (t > 0) ? a2 : nullptr;

        gconv_wmma_kernel<16, 32, 27><<<blocksFor(M0), blk, 0, stream>>>(
            ft, nbr0, p_subm1, bn32 + 0 * 128, nullptr, 0.f, xA, M0);
        gconv_wmma_kernel<32, 32, 27><<<blocksFor(M0), blk, 0, stream>>>(
            xA, nbr0, p_subm2, bn32 + 1 * 128, nullptr, 0.f, xB, M0);
        gconv_wmma_kernel<32, 64, 27><<<blocksFor(M1), blk, 0, stream>>>(
            xB, idx1, p_spcv1, bn64 + 0 * 256, A0, s0scale, x3, M1);
        gconv_wmma_kernel<64, 64, 27><<<blocksFor(M1), blk, 0, stream>>>(
            x3, nbr1, p_subm3, bn64 + 1 * 256, nullptr, 0.f, yA, M1);
        gconv_wmma_kernel<64, 64, 27><<<blocksFor(M1), blk, 0, stream>>>(
            yA, nbr1, p_subm4, bn64 + 2 * 256, nullptr, 0.f, yB, M1);
        gconv_wmma_kernel<64, 64, 27><<<blocksFor(M1), blk, 0, stream>>>(
            yB, nbr1, p_subm5, bn64 + 3 * 256, nullptr, 0.f, yA, M1);  // x6
        gconv_wmma_kernel<64, 64, 27><<<blocksFor(M2), blk, 0, stream>>>(
            yA, idx2, p_spcv2, bn64 + 4 * 256, A1, s1scale, x7, M2);
        gconv_wmma_kernel<64, 64, 27><<<blocksFor(M2), blk, 0, stream>>>(
            x7, nbr2, p_subm6, bn64 + 5 * 256, nullptr, 0.f, zA, M2);
        gconv_wmma_kernel<64, 64, 27><<<blocksFor(M2), blk, 0, stream>>>(
            zA, nbr2, p_subm7, bn64 + 6 * 256, nullptr, 0.f, zB, M2);
        gconv_wmma_kernel<64, 64, 27><<<blocksFor(M2), blk, 0, stream>>>(
            zB, nbr2, p_subm8, bn64 + 7 * 256, nullptr, 0.f, zA, M2);  // x10
        gconv_wmma_kernel<64, 64, 3><<<blocksFor(M3), blk, 0, stream>>>(
            zA, idx3, p_spcv3, bn64 + 8 * 256, A2, s2scale, x11, M3);

        scatter_out_kernel<<<dim3((unsigned)M3), dim3(64), 0, stream>>>(
            x11, coords3, (float*)d_out, M3, t);

        if (t + 1 < T) {
            // next-frame temporal vectors: a = sum_rows(adapt(s)); consumed as a/(10*M)
            zero_f32_kernel<<<dim3(1), dim3(64), 0, stream>>>(a0, 64);
            zero_f32_kernel<<<dim3(1), dim3(64), 0, stream>>>(a1, 64);
            zero_f32_kernel<<<dim3(1), dim3(64), 0, stream>>>(a2, 64);
            adapt_mean_kernel<<<dim3(120), dim3(64), 0, stream>>>(x3,  M1, w_ad1, w_ad2, bn_ad, a0);
            adapt_mean_kernel<<<dim3(120), dim3(64), 0, stream>>>(x7,  M2, w_ad1, w_ad2, bn_ad, a1);
            adapt_mean_kernel<<<dim3(120), dim3(64), 0, stream>>>(x11, M3, w_ad1, w_ad2, bn_ad, a2);
        }
    }
}
